// NHPModel_90623809946311
// MI455X (gfx1250) — matrix operand order repaired and checked by reading, not compiled, not run on hardware
//
#include <hip/hip_runtime.h>

#define S_LEN  1024
#define BATCH  64
#define HDIM   512
#define NGATES 7
#define N7H    (NGATES * HDIM)   // 3584
#define NBLK   16
#define HSLICE 32                // hidden columns per block

typedef __attribute__((ext_vector_type(16))) __bf16 v16bf;
typedef __attribute__((ext_vector_type(8)))  __bf16 v8bf;
typedef __attribute__((ext_vector_type(8)))  float  v8f;
typedef __attribute__((ext_vector_type(4)))  int    v4i;

// ws layout (byte offsets, 256-aligned)
#define WS_EMBW   0            // float [3 * 3584]            = 43008 B
#define WS_PACKB  172032       // __bf16 [7*32*16*512]        = 3670016 B
#define WS_HID    3842048      // __bf16 [2][64*512]          = 131072 B
#define WS_BAR    3973120      // unsigned [2] (+pad)         = 256 B
#define WS_PROJ   3973376      // float [16][64*1024*5]       = 20971520 B

// dynamic LDS layout (byte offsets)
#define LDS_B     0            // __bf16 [192 frags * 512]    = 196608 B
#define LDS_EMBW  196608       // float [3][7][32]            = 2688 B
#define LDS_P     199296       // float [32][5]               = 640 B
#define LDS_HID   199936       // float [64][33]              = 8448 B
#define LDS_EV    208384       // int [64]                    = 256 B
#define LDS_T     208640       // float [64]                  = 256 B
#define LDS_TOTAL 208896

__device__ constexpr int GMAP[6] = {0, 1, 2, 4, 5, 6};  // skip unused forget_target gate

#if __has_builtin(__builtin_amdgcn_global_load_async_to_lds_b128) && \
    __has_builtin(__builtin_amdgcn_s_wait_asynccnt)
#define USE_ASYNC_LDS 1
typedef __attribute__((address_space(1))) v4i as1_v4i;
typedef __attribute__((address_space(3))) v4i as3_v4i;
#else
#define USE_ASYNC_LDS 0
#endif

__device__ __forceinline__ float sigmoid_f(float x) { return 1.0f / (1.0f + __expf(-x)); }
__device__ __forceinline__ float softplus_f(float x) { return x > 20.0f ? x : log1pf(__expf(x)); }

__global__ void zero_u32(unsigned* __restrict__ p, int n) {
  int i = blockIdx.x * blockDim.x + threadIdx.x;
  if (i < n) p[i] = 0u;
}

// embW[e][n] = sum_k emb[e][k] * W[k][n] + b[n]   (top half of W: rows 0..511)
__global__ void embw_kernel(const float* __restrict__ emb, const float* __restrict__ W,
                            const float* __restrict__ b, float* __restrict__ embW) {
  int idx = blockIdx.x * blockDim.x + threadIdx.x;
  if (idx >= 3 * N7H) return;
  int e = idx / N7H, n = idx - e * N7H;
  float acc = b[n];
  for (int k = 0; k < HDIM; ++k) acc += emb[e * HDIM + k] * W[k * N7H + n];
  embW[idx] = acc;
}

// Pack W_bot (rows 512..1023 of W) into bf16 WMMA B-fragments:
// frag f = (g*32 + ht)*16 + kb ; lane l, elem j:
//   K = kb*32 + (l>>4)*16 + j ; N = g*512 + ht*16 + (l&15)
__global__ void packb_kernel(const float* __restrict__ W, __bf16* __restrict__ pb) {
  int idx = blockIdx.x * blockDim.x + threadIdx.x;
  if (idx >= NGATES * 32 * 16 * 512) return;
  int w    = idx & 511;
  int f    = idx >> 9;
  int lane = w >> 4, j = w & 15;
  int kb   = f & 15;
  int fht  = f >> 4;
  int g    = fht >> 5;
  int ht   = fht & 31;
  int K = kb * 32 + (lane >> 4) * 16 + j;
  int N = g * HDIM + ht * 16 + (lane & 15);
  pb[idx] = (__bf16)W[(size_t)(HDIM + K) * N7H + N];
}

__device__ __forceinline__ void grid_barrier(unsigned* cnt, unsigned* gen) {
  __threadfence();          // publish this wave's hidden-state stores device-wide
  __syncthreads();
  if (threadIdx.x == 0) {
    unsigned g = __atomic_load_n(gen, __ATOMIC_RELAXED);
    if (__atomic_fetch_add(cnt, 1u, __ATOMIC_ACQ_REL) == NBLK - 1u) {
      __atomic_store_n(cnt, 0u, __ATOMIC_RELAXED);
      __atomic_store_n(gen, g + 1u, __ATOMIC_RELEASE);
    } else {
      while (__atomic_load_n(gen, __ATOMIC_ACQUIRE) == g) __builtin_amdgcn_s_sleep(2);
    }
  }
  __syncthreads();
  __threadfence();          // acquire: see other blocks' stores
}

__global__ __launch_bounds__(256) void scan_kernel(
    const int* __restrict__ ev_seq, const float* __restrict__ t_seq,
    const float* __restrict__ embW, const __bf16* __restrict__ packedB,
    __bf16* __restrict__ hiddenB, float* __restrict__ proj,
    const float* __restrict__ Wi, const float* __restrict__ Wt,
    const float* __restrict__ We, unsigned* __restrict__ bar) {
  extern __shared__ char smem[];
  __bf16* Bs = (__bf16*)(smem + LDS_B);                               // 192 KB weights
  float (*embW_s)[NGATES][HSLICE] = (float (*)[NGATES][HSLICE])(smem + LDS_EMBW);
  float (*P_s)[5]          = (float (*)[5])(smem + LDS_P);
  float (*hid_s)[HSLICE+1] = (float (*)[HSLICE+1])(smem + LDS_HID);
  int*   ev_s = (int*)(smem + LDS_EV);
  float* T_s  = (float*)(smem + LDS_T);

  const int tid  = threadIdx.x;
  const int blk  = blockIdx.x;
  const int h0   = blk * HSLICE;
  const int wave = tid >> 5;        // 8 wave32 waves
  const int lane = tid & 31;
  const int m0   = (wave & 3) * 16; // M tile: 4 waves cover B=64
  const int nt   = wave >> 2;       // N subtile 0/1 within the 32-wide slice
  const int ll   = lane & 15;
  const int sel  = lane >> 4;

  // ---- one-time fill: this block's 192KB recurrent-weight slice -> LDS ----
  // local frag fl = gi*32 + ntl*16 + kb ; global frag = (GMAP[gi]*32 + blk*2 + ntl)*16 + kb
  for (int i = tid; i < 192 * 64; i += 256) {       // 16B chunks
    int fl = i >> 6;
    int ch = i & 63;
    int gi  = fl >> 5;
    int ntl = (fl >> 4) & 1;
    int kb  = fl & 15;
    size_t gsrc = ((size_t)((GMAP[gi] * 32 + blk * 2 + ntl) * 16 + kb) << 10) + (ch << 4);
    int    loff = LDS_B + fl * 1024 + (ch << 4);
#if USE_ASYNC_LDS
    __builtin_amdgcn_global_load_async_to_lds_b128(
        (as1_v4i*)(const_cast<char*>((const char*)packedB + gsrc)),
        (as3_v4i*)(smem + loff), 0, 0);
#else
    *(uint4*)(smem + loff) = *(const uint4*)((const char*)packedB + gsrc);
#endif
  }
#if USE_ASYNC_LDS
  __builtin_amdgcn_s_wait_asynccnt(0);
#endif

  for (int i = tid; i < 3 * NGATES * HSLICE; i += 256) {
    int e = i / (NGATES * HSLICE);
    int r = i - e * (NGATES * HSLICE);
    int g = r / HSLICE, hh = r - g * HSLICE;
    embW_s[e][g][hh] = embW[e * N7H + g * HDIM + h0 + hh];
  }
  for (int i = tid; i < HSLICE; i += 256) {
    int h = h0 + i;
    P_s[i][0] = Wi[h * 2];
    P_s[i][1] = Wi[h * 2 + 1];
    P_s[i][2] = Wt[h];
    P_s[i][3] = We[h * 2];
    P_s[i][4] = We[h * 2 + 1];
  }
  __syncthreads();

  const v8f zero8 = {0.f, 0.f, 0.f, 0.f, 0.f, 0.f, 0.f, 0.f};
  v8f cd = zero8;                   // cell_decay  (resident in VGPRs for all S steps)
  v8f ct = zero8;                   // cell_target

  const int arow     = m0 + ll;       // A-fragment row per lane
  const int mrow0    = m0 + sel * 8;  // C/D fragment: rows mrow0..mrow0+7
  const int hh_local = nt * 16 + ll;  // column within block's slice
  const int mycol    = h0 + hh_local;
  // per-lane base (elements) into LDS B fragments for this wave's n-subtile
  const int bbase    = nt * (16 << 9) + lane * 16;

  for (int t = 0; t < S_LEN; ++t) {
    if (tid < BATCH) {
      ev_s[tid] = ev_seq[tid * S_LEN + t];
      T_s[tid]  = t_seq[tid * S_LEN + t];
    }
    __syncthreads();

    const __bf16* hin  = hiddenB + (size_t)(t & 1) * (BATCH * HDIM);
    __bf16*       hout = hiddenB + (size_t)((t + 1) & 1) * (BATCH * HDIM);

    v8f acc[6];
    #pragma unroll
    for (int g = 0; g < 6; ++g) acc[g] = zero8;

    #pragma unroll 2
    for (int kb = 0; kb < 16; ++kb) {
      // A fragment (bf16 16x32) from L2-resident ping-pong hidden buffer
      const __bf16* ap = hin + (size_t)arow * HDIM + kb * 32 + sel * 16;
      v8bf alo = *(const v8bf*)ap;
      v8bf ahi = *(const v8bf*)(ap + 8);
      v16bf a = __builtin_shufflevector(alo, ahi, 0, 1, 2, 3, 4, 5, 6, 7,
                                                  8, 9, 10, 11, 12, 13, 14, 15);
      // Stage ALL 6 B fragments in distinct registers first so the 12
      // ds_load_b128 stay in flight, then issue the 6 WMMAs back-to-back.
      v16bf bfr[6];
      #pragma unroll
      for (int gi = 0; gi < 6; ++gi) {
        const __bf16* bp = Bs + ((gi * 32 + kb) << 9) + bbase;
        v8bf blo = *(const v8bf*)bp;
        v8bf bhi = *(const v8bf*)(bp + 8);
        bfr[gi] = __builtin_shufflevector(blo, bhi, 0, 1, 2, 3, 4, 5, 6, 7,
                                                    8, 9, 10, 11, 12, 13, 14, 15);
      }
      #pragma unroll
      for (int gi = 0; gi < 6; ++gi) {
        acc[gi] = __builtin_amdgcn_wmma_f32_16x16x32_bf16(
            false, a, false, bfr[gi], (short)0, acc[gi], false, false);
      }
    }

    // Elementwise CT-LSTM update directly on WMMA fragments (C/D layout:
    // VGPR r, lane l -> M = mrow0 + r, N = ll)
    #pragma unroll
    for (int r = 0; r < 8; ++r) {
      const int   m  = mrow0 + r;
      const int   ev = ev_s[m];
      const float T  = T_s[m];
      float ig  = acc[0][r] + embW_s[ev][0][hh_local];
      float fg  = acc[1][r] + embW_s[ev][1][hh_local];
      float itg = acc[2][r] + embW_s[ev][2][hh_local];
      float og  = acc[3][r] + embW_s[ev][4][hh_local];
      float zg  = acc[4][r] + embW_s[ev][5][hh_local];
      float dg  = acc[5][r] + embW_s[ev][6][hh_local];
      float iv  = sigmoid_f(ig);
      float fv  = sigmoid_f(fg);
      float itv = sigmoid_f(itg);
      float ov  = sigmoid_f(og);
      float zv  = tanhf(zg);
      float dv  = softplus_f(dg);
      float cell = fv * cd[r] + iv * zv;
      float ctn  = itv * ct[r] + itv * zv;
      float cdn  = ctn + (cell - ctn) * __expf(-dv * T);
      float hval = ov * tanhf(cell);
      ct[r] = ctn;
      cd[r] = cdn;
      hout[(size_t)m * HDIM + mycol] = (__bf16)hval;   // bf16 A operand for next step
      hid_s[m][hh_local] = hval;                       // f32 for projections
    }
    __syncthreads();

    // Deterministic per-block partial projections (Wi, Wt, We over this h-slice)
    if (tid < BATCH) {
      float a0 = 0.f, a1 = 0.f, a2 = 0.f, a3 = 0.f, a4 = 0.f;
      #pragma unroll 8
      for (int j = 0; j < HSLICE; ++j) {
        float h = hid_s[tid][j];
        a0 += h * P_s[j][0];
        a1 += h * P_s[j][1];
        a2 += h * P_s[j][2];
        a3 += h * P_s[j][3];
        a4 += h * P_s[j][4];
      }
      float* p = proj + ((size_t)blk * BATCH * S_LEN + (size_t)tid * S_LEN + t) * 5;
      p[0] = a0; p[1] = a1; p[2] = a2; p[3] = a3; p[4] = a4;
    }

    grid_barrier(bar, bar + 1);   // publish hidden(t+1), release next step
  }
}

__global__ void finalize_kernel(const float* __restrict__ proj, const float* __restrict__ bi,
                                float* __restrict__ out) {
  int idx = blockIdx.x * blockDim.x + threadIdx.x;   // (b*S + s)
  if (idx >= BATCH * S_LEN) return;
  float s0 = 0.f, s1 = 0.f, s2 = 0.f, s3 = 0.f, s4 = 0.f;
  for (int blk = 0; blk < NBLK; ++blk) {
    const float* p = proj + ((size_t)blk * BATCH * S_LEN + idx) * 5;
    s0 += p[0]; s1 += p[1]; s2 += p[2]; s3 += p[3]; s4 += p[4];
  }
  out[idx * 2]     = softplus_f(s0 + bi[0]);                 // intensity [B,S,E]
  out[idx * 2 + 1] = softplus_f(s1 + bi[1]);
  out[BATCH * S_LEN * 2 + idx] = s2;                         // time_pred [B,S]
  out[BATCH * S_LEN * 3 + idx * 2]     = s3;                 // event_pred [B,S,E]
  out[BATCH * S_LEN * 3 + idx * 2 + 1] = s4;
}

extern "C" void kernel_launch(void* const* d_in, const int* in_sizes, int n_in,
                              void* d_out, int out_size, void* d_ws, size_t ws_size,
                              hipStream_t stream) {
  const int*   ev  = (const int*)d_in[0];    // event_seq [64,1024] i32
  const float* ts  = (const float*)d_in[1];  // time_seq  [64,1024] f32
  const float* emb = (const float*)d_in[2];  // [3,512]
  const float* W   = (const float*)d_in[3];  // [1024,3584]
  const float* b   = (const float*)d_in[4];  // [3584]
  const float* Wi  = (const float*)d_in[5];  // [512,2]
  const float* bi  = (const float*)d_in[6];  // [2]
  const float* Wt  = (const float*)d_in[7];  // [512,1]
  const float* We  = (const float*)d_in[8];  // [512,2]
  float* out = (float*)d_out;

  char* ws = (char*)d_ws;
  float*    embW    = (float*)(ws + WS_EMBW);
  __bf16*   packedB = (__bf16*)(ws + WS_PACKB);
  __bf16*   hiddenB = (__bf16*)(ws + WS_HID);
  unsigned* bar     = (unsigned*)(ws + WS_BAR);
  float*    proj    = (float*)(ws + WS_PROJ);

  // allow >64KB dynamic LDS (CDNA5 WGP has 320KB)
  (void)hipFuncSetAttribute((const void*)scan_kernel,
                            hipFuncAttributeMaxDynamicSharedMemorySize, LDS_TOTAL);

  // zero hidden ping-pong + barrier words (contiguous region) every call
  int nz = (131072 + 256) / 4;
  zero_u32<<<(nz + 255) / 256, 256, 0, stream>>>((unsigned*)(ws + WS_HID), nz);
  embw_kernel<<<(3 * N7H + 255) / 256, 256, 0, stream>>>(emb, W, b, embW);
  packb_kernel<<<(NGATES * 32 * 16 * 512 + 255) / 256, 256, 0, stream>>>(W, packedB);
  scan_kernel<<<NBLK, 256, LDS_TOTAL, stream>>>(ev, ts, embW, packedB, hiddenB, proj,
                                                Wi, Wt, We, bar);
  finalize_kernel<<<(BATCH * S_LEN + 255) / 256, 256, 0, stream>>>(proj, bi, out);
}
